// GCN_55301998903731
// MI455X (gfx1250) — matrix operand order repaired and checked by reading, not compiled, not run on hardware
//
#include <hip/hip_runtime.h>
#include <hip/hip_bf16.h>
#include <math.h>

typedef __attribute__((ext_vector_type(16))) __bf16 v16bf;
typedef __attribute__((ext_vector_type(8)))  float  v8f;

#define NFEAT 128
#define NHID  128

static __device__ __forceinline__ unsigned short f2bf(float f) {
  unsigned u = __float_as_uint(f);
  unsigned r = u + 0x7FFFu + ((u >> 16) & 1u);   // round to nearest even
  return (unsigned short)(r >> 16);
}

// ---------------------------------------------------------------- conversions
__global__ __launch_bounds__(256) void k_f32_to_bf16(const float* __restrict__ in,
                                                     unsigned short* __restrict__ out,
                                                     int n) {
  int i = blockIdx.x * 256 + threadIdx.x;
  if (i < n) out[i] = f2bf(in[i]);
}

// Pack W[K=128, N] (row-major f32) into WMMA B-fragment order (bf16):
// Bp[((nt*KB + kb)*32 + lane)*16 + i] = W[(kb*32 + (i>>3)*16 + (lane>>4)*8 + (i&7))*N + nt*16 + (lane&15)]
__global__ __launch_bounds__(256) void k_pack_w(const float* __restrict__ W,
                                                unsigned short* __restrict__ Bp,
                                                int N, int KB) {
  int idx = blockIdx.x * 256 + threadIdx.x;
  if (idx >= 128 * N) return;
  int i    = idx & 15;
  int lane = (idx >> 4) & 31;
  int blk  = idx >> 9;
  int kb   = blk % KB;
  int nt   = blk / KB;
  int col  = nt * 16 + (lane & 15);
  int k    = kb * 32 + (i >> 3) * 16 + (lane >> 4) * 8 + (i & 7);
  Bp[idx] = f2bf(W[k * N + col]);
}

// ---------------------------------------------------------------- WMMA GEMM
// C[M,N] (f32, row-major) = A[M,128] (bf16, row-major) @ Bp (packed fragments)
union Frag16 { uint4 q[2]; v16bf v; };

template <int NT>
__global__ __launch_bounds__(256) void k_gemm_bf16(const unsigned short* __restrict__ A,
                                                   const unsigned short* __restrict__ Bp,
                                                   float* __restrict__ C, int M) {
  constexpr int K = 128, KB = 4, N = NT * 16;
  const int lane  = threadIdx.x & 31;
  const int wave  = threadIdx.x >> 5;
  const int mtile = blockIdx.x * 8 + wave;
  if (mtile * 16 >= M) return;                 // wave-uniform: EXEC stays all-ones

  // A fragment loads per ISA 16x32 bf16 A layout:
  // lanes 0-15: row=lane,   K = kb*32 + {0..7} and {16..23}
  // lanes16-31: row=lane-16,K = kb*32 + {8..15} and {24..31}
  const int row = mtile * 16 + (lane & 15);
  const int ko  = (lane >> 4) * 8;
  const unsigned short* arow = A + (size_t)row * K;

  Frag16 a[KB];
#pragma unroll
  for (int kb = 0; kb < KB; ++kb) {
    a[kb].q[0] = *(const uint4*)(arow + kb * 32 + ko);
    a[kb].q[1] = *(const uint4*)(arow + kb * 32 + 16 + ko);
  }

#pragma unroll
  for (int nt = 0; nt < NT; ++nt) {
    v8f acc = {};
#pragma unroll
    for (int kb = 0; kb < KB; ++kb) {
      Frag16 b;
      const uint4* bp = (const uint4*)(Bp + ((size_t)(nt * KB + kb) * 32 + lane) * 16);
      b.q[0] = bp[0];
      b.q[1] = bp[1];
      acc = __builtin_amdgcn_wmma_f32_16x16x32_bf16(false, a[kb].v, false, b.v,
                                                    (short)0, acc, false, false);
    }
    // C/D layout: element (vgpr r, lane l) -> row = r + 8*(l>>4), col = l&15
    float* cb = C + (size_t)mtile * 16 * N + nt * 16 + (lane & 15);
    const int rb = (lane >> 4) * 8;
#pragma unroll
    for (int r = 0; r < 8; ++r)
      cb[(size_t)(rb + r) * N] = acc[r];
  }
}

// ---------------------------------------------------------------- SpMM scatter
// agg[dst[e], :] += support[src[e], :] * w[e]   (one wave per edge, L2-resident)
__global__ __launch_bounds__(256) void k_spmm_scatter(const float* __restrict__ S,
                                                      const int* __restrict__ src,
                                                      const int* __restrict__ dst,
                                                      const float* __restrict__ w,
                                                      float* __restrict__ agg,
                                                      int E, int F) {
  const int lane = threadIdx.x & 31;
  const int gw   = (blockIdx.x * 256 + threadIdx.x) >> 5;
  const int nw   = (gridDim.x * 256) >> 5;
  for (int e = gw; e < E; e += nw) {
    const int en = e + nw;
    if (en < E && lane == 0)
      __builtin_prefetch(S + (size_t)src[en] * F, 0, 1);  // global_prefetch_b8
    const int   s  = src[e];
    const int   d  = dst[e];
    const float we = w[e];
    const float4* srow = (const float4*)(S + (size_t)s * F);
    float*        drow = agg + (size_t)d * F;
    for (int c4 = lane; c4 < (F >> 2); c4 += 32) {
      float4 v = srow[c4];
      float* p = drow + c4 * 4;
      unsafeAtomicAdd(p + 0, v.x * we);
      unsafeAtomicAdd(p + 1, v.y * we);
      unsafeAtomicAdd(p + 2, v.z * we);
      unsafeAtomicAdd(p + 3, v.w * we);
    }
  }
}

// -------------------------------------------- fused bias + relu -> bf16 (h1)
__global__ __launch_bounds__(256) void k_bias_relu_bf16(const float* __restrict__ a,
                                                        const float* __restrict__ b,
                                                        unsigned short* __restrict__ outbf,
                                                        size_t total, int fmask) {
  size_t i = (size_t)blockIdx.x * 256 + threadIdx.x;
  if (i < total) {
    float v = a[i] + b[i & fmask];
    outbf[i] = f2bf(v > 0.f ? v : 0.f);
  }
}

// --------------------------------- fused bias + relu + column-sum pool (h2)
// g[col] += sum_r relu(agg[r,col] + bias[col])   (h2 never materialized)
__global__ __launch_bounds__(256) void k_colsum_bias_relu(const float* __restrict__ agg,
                                                          const float* __restrict__ bias,
                                                          float* __restrict__ g,
                                                          int Nn, int F) {
  const int col = threadIdx.x;                 // blockDim.x == F == 256
  const float b = bias[col];
  float acc = 0.f;
  for (int r = blockIdx.x; r < Nn; r += gridDim.x) {
    float v = agg[(size_t)r * F + col] + b;
    acc += v > 0.f ? v : 0.f;
  }
  unsafeAtomicAdd(&g[col], acc);
}

// ---------------------------------------------------------------- MLP head
__global__ __launch_bounds__(256) void k_head(const float* __restrict__ gsum,
                                              const float* __restrict__ W3,
                                              const float* __restrict__ b3,
                                              const float* __restrict__ W4,
                                              const float* __restrict__ b4,
                                              float* __restrict__ out) {
  __shared__ float g[256];
  __shared__ float h[128];
  __shared__ float lg[10];
  const int t = threadIdx.x;
  float v = gsum[t];
  g[t] = v > 0.f ? v : 0.f;                    // relu(pool)
  __syncthreads();
  if (t < 128) {
    float acc = b3[t];
    for (int k = 0; k < 256; ++k) acc = fmaf(g[k], W3[k * 128 + t], acc);
    h[t] = acc > 0.f ? acc : 0.f;
  }
  __syncthreads();
  if (t < 10) {
    float acc = b4[t];
    for (int j = 0; j < 128; ++j) acc = fmaf(h[j], W4[j * 10 + t], acc);
    lg[t] = acc;
  }
  __syncthreads();
  if (t == 0) {
    float m = lg[0];
    for (int c = 1; c < 10; ++c) m = fmaxf(m, lg[c]);
    float s = 0.f;
    for (int c = 0; c < 10; ++c) s += __expf(lg[c] - m);
    float ls = __logf(s);
    for (int c = 0; c < 10; ++c) out[c] = lg[c] - m - ls;
  }
}

// ---------------------------------------------------------------- launch
extern "C" void kernel_launch(void* const* d_in, const int* in_sizes, int n_in,
                              void* d_out, int out_size, void* d_ws, size_t ws_size,
                              hipStream_t stream) {
  const float* x   = (const float*)d_in[0];
  const int*   es  = (const int*)d_in[1];
  const int*   ed  = (const int*)d_in[2];
  const float* ew  = (const float*)d_in[3];
  const float* W1  = (const float*)d_in[4];
  const float* b1  = (const float*)d_in[5];
  const float* W2  = (const float*)d_in[6];
  const float* b2  = (const float*)d_in[7];
  const float* W3  = (const float*)d_in[8];
  const float* b3  = (const float*)d_in[9];
  const float* W4  = (const float*)d_in[10];
  const float* b4  = (const float*)d_in[11];
  float* out = (float*)d_out;

  const int Nn = in_sizes[0] / NFEAT;          // 50000
  const int E  = in_sizes[1];                  // 800000
  const int F1 = NFEAT;                        // 128
  const int F2 = 2 * NHID;                     // 256

  char* ws = (char*)d_ws;
  size_t off = 0;
  unsigned short* abf = (unsigned short*)(ws + off); off += (size_t)Nn * F1 * 2; // x / h1 bf16
  unsigned short* W1p = (unsigned short*)(ws + off); off += (size_t)F1 * F1 * 2;
  unsigned short* W2p = (unsigned short*)(ws + off); off += (size_t)F1 * F2 * 2;
  float* bufA = (float*)(ws + off); off += (size_t)Nn * F2 * 4;                  // support1 / agg2
  float* bufB = (float*)(ws + off); off += (size_t)Nn * F2 * 4;                  // agg1 / support2
  float* g0   = (float*)(ws + off); off += 1024;
  (void)n_in; (void)out_size; (void)ws_size;

  const int mt = (Nn + 15) / 16;

  // layer 1: support1 = x @ W1
  k_f32_to_bf16<<<(Nn * F1 + 255) / 256, 256, 0, stream>>>(x, abf, Nn * F1);
  k_pack_w<<<(F1 * F1 + 255) / 256, 256, 0, stream>>>(W1, W1p, F1, 4);
  k_pack_w<<<(F1 * F2 + 255) / 256, 256, 0, stream>>>(W2, W2p, F2, 4);
  k_gemm_bf16<8><<<(mt + 7) / 8, 256, 0, stream>>>(abf, W1p, bufA, Nn);

  // agg1 = scatter-add over edges; h1 = relu(agg1 + b1) -> bf16 directly
  hipMemsetAsync(bufB, 0, (size_t)Nn * F1 * 4, stream);
  k_spmm_scatter<<<12500, 256, 0, stream>>>(bufA, es, ed, ew, bufB, E, F1);
  {
    size_t tot = (size_t)Nn * F1;
    k_bias_relu_bf16<<<(int)((tot + 255) / 256), 256, 0, stream>>>(bufB, b1, abf, tot, F1 - 1);
  }

  // layer 2: support2 = h1 @ W2 (bufB reused; f32 agg1 dead)
  k_gemm_bf16<16><<<(mt + 7) / 8, 256, 0, stream>>>(abf, W2p, bufB, Nn);

  // agg2 = scatter-add; pool = sum relu(agg2 + b2) fused (h2 never written)
  hipMemsetAsync(bufA, 0, (size_t)Nn * F2 * 4, stream);
  k_spmm_scatter<<<12500, 256, 0, stream>>>(bufB, es, ed, ew, bufA, E, F2);
  hipMemsetAsync(g0, 0, 1024, stream);
  k_colsum_bias_relu<<<1024, 256, 0, stream>>>(bufA, b2, g0, Nn, F2);

  // head
  k_head<<<1, 256, 0, stream>>>(g0, W3, b3, W4, b4, out);
}
